// HeFormer_89644557402716
// MI455X (gfx1250) — matrix operand order repaired and checked by reading, not compiled, not run on hardware
//
#include <hip/hip_runtime.h>
#include <hip/hip_bf16.h>

#define NN 100000
#define FF 512
#define HH 128
#define EE 1600000
#define OC 64
#define EPSF 1e-5f

typedef __attribute__((ext_vector_type(16))) __bf16 v16bf;
typedef __attribute__((ext_vector_type(8)))  float  v8f;

__device__ __forceinline__ __bf16 f2bf(float f) {
    union { float f; unsigned u; } a; a.f = f;
    unsigned r = a.u + 0x7fffu + ((a.u >> 16) & 1u);
    union { unsigned short s; __bf16 b; } c; c.s = (unsigned short)(r >> 16);
    return c.b;
}

union AFrag { unsigned u[8]; v16bf v; };

// ---------------------------------------------------------------------------
// Generic bf16 WMMA GEMM:  C[r, n0..n0+63] = A[r, 0..K) * Bt[n][k] + bias[n]
// A row-major [rows, lda] bf16; Bt = B transposed [M, ldb] bf16; C f32 [rows, ldc].
// 1 wave => 16 rows x 64 cols. blockDim 256 (8 waves). grid.y covers M/64.
// ---------------------------------------------------------------------------
__global__ __launch_bounds__(256) void gemm_bf16(
    const __bf16* __restrict__ A, int lda,
    const __bf16* __restrict__ Bt, int ldb,
    const float* __restrict__ bias,
    float* __restrict__ Cout, int ldc,
    int strips, int K)
{
    const int wave  = threadIdx.x >> 5;
    const int lane  = threadIdx.x & 31;
    const int strip = blockIdx.x * 8 + wave;
    if (strip >= strips) return;                 // wave-uniform: EXEC all-ones below
    const int m0   = strip << 4;
    const int n0   = blockIdx.y << 6;
    const int hi   = lane >> 4;                  // lane>=16 ?
    const int arow = m0 + (lane & 15);
    const int bcol = n0 + (lane & 15);

    v8f c0 = {}, c1 = {}, c2 = {}, c3 = {};
    for (int k = 0; k < K; k += 32) {
        // A fragment 16x32: lane<16 K in {0..15}; lane>=16 K in {16..31}
        AFrag a;
        const unsigned* pa0 = (const unsigned*)(A + (size_t)arow * lda + k + hi * 16);
        const unsigned* pa1 = (const unsigned*)(A + (size_t)arow * lda + k + hi * 16 + 8);
        a.u[0] = pa0[0]; a.u[1] = pa0[1]; a.u[2] = pa0[2]; a.u[3] = pa0[3];
        a.u[4] = pa1[0]; a.u[5] = pa1[1]; a.u[6] = pa1[2]; a.u[7] = pa1[3];
        const int kb = k + hi * 16;              // B fragment 32x16: lanes>=16 hold K=16..31
        {
            AFrag b; const unsigned* pb = (const unsigned*)(Bt + (size_t)(bcol +  0) * ldb + kb);
            #pragma unroll
            for (int v = 0; v < 8; ++v) b.u[v] = pb[v];
            c0 = __builtin_amdgcn_wmma_f32_16x16x32_bf16(false, a.v, false, b.v, (short)0, c0, false, false);
        }
        {
            AFrag b; const unsigned* pb = (const unsigned*)(Bt + (size_t)(bcol + 16) * ldb + kb);
            #pragma unroll
            for (int v = 0; v < 8; ++v) b.u[v] = pb[v];
            c1 = __builtin_amdgcn_wmma_f32_16x16x32_bf16(false, a.v, false, b.v, (short)0, c1, false, false);
        }
        {
            AFrag b; const unsigned* pb = (const unsigned*)(Bt + (size_t)(bcol + 32) * ldb + kb);
            #pragma unroll
            for (int v = 0; v < 8; ++v) b.u[v] = pb[v];
            c2 = __builtin_amdgcn_wmma_f32_16x16x32_bf16(false, a.v, false, b.v, (short)0, c2, false, false);
        }
        {
            AFrag b; const unsigned* pb = (const unsigned*)(Bt + (size_t)(bcol + 48) * ldb + kb);
            #pragma unroll
            for (int v = 0; v < 8; ++v) b.u[v] = pb[v];
            c3 = __builtin_amdgcn_wmma_f32_16x16x32_bf16(false, a.v, false, b.v, (short)0, c3, false, false);
        }
    }
    // C layout: VGPR r -> row m0+r (lane<16) / m0+8+r (lane>=16), col = n0 + lane&15
    const int rbase = m0 + hi * 8;
    const int cb    = n0 + (lane & 15);
    float b0 = 0.f, b1 = 0.f, b2 = 0.f, b3 = 0.f;
    if (bias) { b0 = bias[cb]; b1 = bias[cb + 16]; b2 = bias[cb + 32]; b3 = bias[cb + 48]; }
    #pragma unroll
    for (int r = 0; r < 8; ++r) {
        float* o = Cout + (size_t)(rbase + r) * ldc + cb;
        o[0]  = c0[r] + b0;
        o[16] = c1[r] + b1;
        o[32] = c2[r] + b2;
        o[48] = c3[r] + b3;
    }
}

// ---------------------------------------------------------------------------
// Small helpers
// ---------------------------------------------------------------------------
__global__ void zero_f32_kernel(float* p, long n) {
    for (long i = (long)blockIdx.x * blockDim.x + threadIdx.x; i < n;
         i += (long)gridDim.x * blockDim.x) p[i] = 0.f;
}

__global__ void convert_bf16_kernel(const float* __restrict__ src, __bf16* __restrict__ dst, long n) {
    for (long i = (long)blockIdx.x * blockDim.x + threadIdx.x; i < n;
         i += (long)gridDim.x * blockDim.x) dst[i] = f2bf(src[i]);
}

// qbf[n, 0..255] = bf16(qkv[n, 0..255])  (q part of packed qkv rows of 768)
__global__ void qbf_kernel(const float* __restrict__ qkv, __bf16* __restrict__ qbf, long n) {
    for (long i = (long)blockIdx.x * blockDim.x + threadIdx.x; i < n;
         i += (long)gridDim.x * blockDim.x) {
        long row = i >> 8; int c = (int)(i & 255);
        qbf[i] = f2bf(qkv[row * 768 + c]);
    }
}

// Weight prep: f32 [K,M] -> bf16 transposed [M,K]; biases concatenated.
__global__ void prep_weights_kernel(
    const float* t_fc_w, const float* g_fc_w,
    const float* wq_w, const float* wk_w, const float* wv_w,
    const float* conv_w, const float* fc_w,
    const float* t_fc_b, const float* g_fc_b,
    const float* wq_b, const float* wk_b, const float* wv_b,
    __bf16* wcat, __bf16* wqkv, __bf16* wconv, __bf16* wfc,
    float* bcat, float* bqkv)
{
    const int S0 = 256 * 512, S1 = 768 * 128, S2 = 128 * 128, S3 = 64 * 128;
    const int TOT = S0 + S1 + S2 + S3 + 256 + 768;
    for (int i = blockIdx.x * blockDim.x + threadIdx.x; i < TOT; i += gridDim.x * blockDim.x) {
        if (i < S0) {
            int col = i / 512, k = i % 512;
            float w = (col < 128) ? t_fc_w[k * 128 + col] : g_fc_w[k * 128 + (col - 128)];
            wcat[i] = f2bf(w);
        } else if (i < S0 + S1) {
            int j = i - S0, col = j / 128, k = j % 128;
            float w = (col < 256) ? wq_w[k * 256 + col]
                    : (col < 512) ? wk_w[k * 256 + col - 256]
                                  : wv_w[k * 256 + col - 512];
            wqkv[j] = f2bf(w);
        } else if (i < S0 + S1 + S2) {
            int j = i - S0 - S1, col = j / 128, k = j % 128;
            wconv[j] = f2bf(conv_w[k * 128 + col]);
        } else if (i < S0 + S1 + S2 + S3) {
            int j = i - S0 - S1 - S2, col = j / 128, k = j % 128;
            wfc[j] = f2bf(fc_w[k * 64 + col]);
        } else if (i < S0 + S1 + S2 + S3 + 256) {
            int c = i - S0 - S1 - S2 - S3;
            bcat[c] = (c < 128) ? t_fc_b[c] : g_fc_b[c - 128];
        } else {
            int c = i - S0 - S1 - S2 - S3 - 256;
            bqkv[c] = (c < 256) ? wq_b[c] : (c < 512) ? wk_b[c - 256] : wv_b[c - 512];
        }
    }
}

// LayerNorm0 + ReLU on t-half of pre1 [N,256]; one block per node (128 threads)
__global__ __launch_bounds__(128) void ln0_relu_kernel(
    const float* __restrict__ pre, const float* __restrict__ g, const float* __restrict__ b,
    float* __restrict__ ht, __bf16* __restrict__ htbf)
{
    const int n = blockIdx.x, d = threadIdx.x;
    __shared__ float red[128];
    float v = pre[(size_t)n * 256 + d];
    red[d] = v; __syncthreads();
    for (int s = 64; s; s >>= 1) { if (d < s) red[d] += red[d + s]; __syncthreads(); }
    float mu = red[0] * (1.f / 128.f); __syncthreads();
    float c = v - mu;
    red[d] = c * c; __syncthreads();
    for (int s = 64; s; s >>= 1) { if (d < s) red[d] += red[d + s]; __syncthreads(); }
    float var = red[0] * (1.f / 128.f);
    float y = fmaxf(c * rsqrtf(var + EPSF) * g[d] + b[d], 0.f);
    ht[(size_t)n * 128 + d] = y;
    htbf[(size_t)n * 128 + d] = f2bf(y);
}

// BatchNorm0(eval) + ReLU on g-half of pre1
__global__ void bn0_relu_kernel(const float* __restrict__ pre, const float* __restrict__ g,
                                const float* __restrict__ b, float* __restrict__ hg, long n)
{
    const float sc = rsqrtf(1.f + EPSF);
    for (long i = (long)blockIdx.x * blockDim.x + threadIdx.x; i < n;
         i += (long)gridDim.x * blockDim.x) {
        long row = i >> 7; int d = (int)(i & 127);
        hg[i] = fmaxf(pre[row * 256 + 128 + d] * sc * g[d] + b[d], 0.f);
    }
}

// sum of squares of q and k parts -> sums[0], sums[1]
__global__ void sumsq_kernel(const float* __restrict__ qkv, float* __restrict__ sums, long total) {
    float aq = 0.f, ak = 0.f;
    for (long i = (long)blockIdx.x * blockDim.x + threadIdx.x; i < total;
         i += (long)gridDim.x * blockDim.x) {
        long row = i >> 8; int c = (int)(i & 255);
        float q = qkv[row * 768 + c];        aq += q * q;
        float k = qkv[row * 768 + 256 + c];  ak += k * k;
    }
    __shared__ float rq[256], rk[256];
    rq[threadIdx.x] = aq; rk[threadIdx.x] = ak; __syncthreads();
    for (int s = 128; s; s >>= 1) {
        if (threadIdx.x < s) { rq[threadIdx.x] += rq[threadIdx.x + s]; rk[threadIdx.x] += rk[threadIdx.x + s]; }
        __syncthreads();
    }
    if (!threadIdx.x) { atomicAdd(&sums[0], rq[0]); atomicAdd(&sums[1], rk[0]); }
}

// column sums of raw k: ksum[col] = sum_n qkv[n, 256+col]; one block per col
__global__ __launch_bounds__(256) void ksum_kernel(const float* __restrict__ qkv, float* __restrict__ ksum) {
    const int col = blockIdx.x;
    float acc = 0.f;
    for (int n = threadIdx.x; n < NN; n += 256) acc += qkv[(size_t)n * 768 + 256 + col];
    __shared__ float red[256];
    red[threadIdx.x] = acc; __syncthreads();
    for (int s = 128; s; s >>= 1) { if (threadIdx.x < s) red[threadIdx.x] += red[threadIdx.x + s]; __syncthreads(); }
    if (!threadIdx.x) ksum[col] = red[0];
}

// kvs[h][m][d] += sum_l k[l,h,m]*v[l,h,d]  (split-L, LDS-staged, f32 atomics)
// grid: (64 tiles of 16x16, 2 heads, L-chunks); block 256
__global__ __launch_bounds__(256) void kvs_partial_kernel(
    const float* __restrict__ qkv, float* __restrict__ kvs, int chunk)
{
    const int tile = blockIdx.x, h = blockIdx.y, ch = blockIdx.z;
    const int m0 = (tile >> 3) * 16, d0 = (tile & 7) * 16;
    const int tm = threadIdx.x >> 4, td = threadIdx.x & 15;
    int l0 = ch * chunk, l1 = l0 + chunk; if (l1 > NN) l1 = NN;
    __shared__ float lk[32][16], lv[32][16];
    float acc = 0.f;
    for (int l = l0; l < l1; l += 32) {
        for (int i = threadIdx.x; i < 512; i += 256) {
            int ll = i >> 4, mm = i & 15, lg = l + ll;
            float kk = 0.f, vv = 0.f;
            if (lg < l1) {
                kk = qkv[(size_t)lg * 768 + 256 + h * 128 + m0 + mm];
                vv = qkv[(size_t)lg * 768 + 512 + h * 128 + d0 + mm];
            }
            lk[ll][mm] = kk; lv[ll][mm] = vv;
        }
        __syncthreads();
        #pragma unroll
        for (int j = 0; j < 32; ++j) acc += lk[j][tm] * lv[j][td];
        __syncthreads();
    }
    atomicAdd(&kvs[h * 16384 + (m0 + tm) * 128 + d0 + td], acc);
}

// kvsbf[h][d][m] = bf16(kvs[h][m][d])  (transpose for B operand)
__global__ void kvs_t_bf_kernel(const float* __restrict__ kvs, __bf16* __restrict__ out) {
    for (int i = blockIdx.x * blockDim.x + threadIdx.x; i < 2 * 128 * 128; i += gridDim.x * blockDim.x) {
        int h = i >> 14, d = (i >> 7) & 127, m = i & 127;
        out[i] = f2bf(kvs[h * 16384 + m * 128 + d]);
    }
}

// per-node attention epilogue: scale, +N*v, normalize, head mean, residual, LN1, ReLU
__global__ __launch_bounds__(128) void attn_post_kernel(
    const float* __restrict__ qkv, const float* __restrict__ attnum,
    const float* __restrict__ ht, const float* __restrict__ ksum,
    const float* __restrict__ sums, const float* __restrict__ g,
    const float* __restrict__ b, float* __restrict__ x1)
{
    const int n = blockIdx.x, d = threadIdx.x;
    __shared__ float red[128];
    const float q0 = qkv[(size_t)n * 768 + d];
    const float q1 = qkv[(size_t)n * 768 + 128 + d];
    red[d] = q0 * ksum[d]; __syncthreads();
    for (int s = 64; s; s >>= 1) { if (d < s) red[d] += red[d + s]; __syncthreads(); }
    float dot0 = red[0]; __syncthreads();
    red[d] = q1 * ksum[128 + d]; __syncthreads();
    for (int s = 64; s; s >>= 1) { if (d < s) red[d] += red[d + s]; __syncthreads(); }
    float dot1 = red[0]; __syncthreads();
    const float sc = rsqrtf(sums[0]) * rsqrtf(sums[1]);   // 1/(|q|_F * |k|_F)
    const float fN = (float)NN;
    const float v0 = qkv[(size_t)n * 768 + 512 + d];
    const float v1 = qkv[(size_t)n * 768 + 640 + d];
    const float a0 = attnum[(size_t)n * 256 + d] * sc + fN * v0;
    const float a1 = attnum[(size_t)n * 256 + 128 + d] * sc + fN * v1;
    const float attn = 0.5f * (a0 / (dot0 * sc + fN) + a1 / (dot1 * sc + fN));
    const float h = 0.5f * (attn + ht[(size_t)n * 128 + d]);
    red[d] = h; __syncthreads();
    for (int s = 64; s; s >>= 1) { if (d < s) red[d] += red[d + s]; __syncthreads(); }
    float mu = red[0] * (1.f / 128.f); __syncthreads();
    float c = h - mu;
    red[d] = c * c; __syncthreads();
    for (int s = 64; s; s >>= 1) { if (d < s) red[d] += red[d + s]; __syncthreads(); }
    float var = red[0] * (1.f / 128.f);
    x1[(size_t)n * 128 + d] = fmaxf(c * rsqrtf(var + EPSF) * g[d] + b[d], 0.f);
}

__global__ void deg_kernel(const int* __restrict__ ei, float* __restrict__ deg) {
    int e = blockIdx.x * blockDim.x + threadIdx.x;
    if (e < EE) atomicAdd(&deg[ei[EE + e]], 1.f);
}

// agg[col] += val(e) * hg[row]; one thread = (edge, 4-channel group)
__global__ void scatter_kernel(const int* __restrict__ ei, const float* __restrict__ hg,
                               const float* __restrict__ deg, float* __restrict__ agg)
{
    long tid = (long)blockIdx.x * blockDim.x + threadIdx.x;
    if (tid >= (long)EE * 32) return;
    int e = (int)(tid >> 5), c4 = (int)(tid & 31) * 4;
    int r = ei[e], c = ei[EE + e];
    float dr = deg[r], dc = deg[c];
    float val = (dr > 0.f && dc > 0.f) ? rsqrtf(dr) * rsqrtf(dc) : 0.f;
    const float4 hv = *(const float4*)(hg + (size_t)r * 128 + c4);
    float* dst = agg + (size_t)c * 128 + c4;
    atomicAdd(dst + 0, val * hv.x);
    atomicAdd(dst + 1, val * hv.y);
    atomicAdd(dst + 2, val * hv.z);
    atomicAdd(dst + 3, val * hv.w);
}

// x2 = relu(BN1(h2pre)) + hg; obbf = bf16(0.8*x2 + 0.2*x1)
__global__ void gc_combine_kernel(const float* __restrict__ h2pre, const float* __restrict__ hg,
                                  const float* __restrict__ g, const float* __restrict__ b,
                                  const float* __restrict__ x1, __bf16* __restrict__ obbf, long n)
{
    const float sc = rsqrtf(1.f + EPSF);
    for (long i = (long)blockIdx.x * blockDim.x + threadIdx.x; i < n;
         i += (long)gridDim.x * blockDim.x) {
        int d = (int)(i & 127);
        float y = fmaxf(h2pre[i] * sc * g[d] + b[d], 0.f) + hg[i];
        obbf[i] = f2bf(0.8f * y + 0.2f * x1[i]);
    }
}

// ---------------------------------------------------------------------------
extern "C" void kernel_launch(void* const* d_in, const int* in_sizes, int n_in,
                              void* d_out, int out_size, void* d_ws, size_t ws_size,
                              hipStream_t stream)
{
    const float* x       = (const float*)d_in[0];
    const int*   ei      = (const int*)  d_in[1];
    const float* t_fc_w  = (const float*)d_in[2];
    const float* t_fc_b  = (const float*)d_in[3];
    const float* ln0_g   = (const float*)d_in[4];
    const float* ln0_b   = (const float*)d_in[5];
    const float* wq_w    = (const float*)d_in[6];
    const float* wq_b    = (const float*)d_in[7];
    const float* wk_w    = (const float*)d_in[8];
    const float* wk_b    = (const float*)d_in[9];
    const float* wv_w    = (const float*)d_in[10];
    const float* wv_b    = (const float*)d_in[11];
    const float* ln1_g   = (const float*)d_in[12];
    const float* ln1_b   = (const float*)d_in[13];
    const float* g_fc_w  = (const float*)d_in[14];
    const float* g_fc_b  = (const float*)d_in[15];
    const float* bn0_g   = (const float*)d_in[16];
    const float* bn0_b   = (const float*)d_in[17];
    const float* conv_w  = (const float*)d_in[18];
    const float* conv_b  = (const float*)d_in[19];
    const float* bn1_g   = (const float*)d_in[20];
    const float* bn1_b   = (const float*)d_in[21];
    const float* fc_w    = (const float*)d_in[22];
    const float* fc_b    = (const float*)d_in[23];
    float* out = (float*)d_out;

    // ---- workspace layout (with aliasing; launch order keeps regions disjoint in time)
    char* w = (char*)d_ws;
    size_t off = 0;
    auto take = [&](size_t bytes) { size_t a = off; off += (bytes + 255) & ~(size_t)255; return a; };
    size_t o_r1   = take((size_t)NN * FF * 2);    // xbf [N,512]bf16  -> later attnum [N,256]f32
    size_t o_pre1 = take((size_t)NN * 256 * 4);   // pre1 -> later {qbf, aggbf, obbf}
    size_t o_qkv  = take((size_t)NN * 768 * 4);   // qkv  -> later {agg, h2pre}
    size_t o_ht   = take((size_t)NN * HH * 4);
    size_t o_htbf = take((size_t)NN * HH * 2);
    size_t o_x1   = take((size_t)NN * HH * 4);
    size_t o_hg   = take((size_t)NN * HH * 4);
    size_t o_deg  = take((size_t)NN * 4);
    size_t o_wcat = take((size_t)256 * 512 * 2);
    size_t o_wqkv = take((size_t)768 * 128 * 2);
    size_t o_wcv  = take((size_t)128 * 128 * 2);
    size_t o_wfc  = take((size_t)64 * 128 * 2);
    size_t o_bcat = take(256 * 4);
    size_t o_bqkv = take(768 * 4);
    size_t o_kvs  = take((size_t)2 * 128 * 128 * 4);
    size_t o_kvsb = take((size_t)2 * 128 * 128 * 2);
    size_t o_ksum = take(256 * 4);
    size_t o_sums = take(2 * 4);

    __bf16* xbf    = (__bf16*)(w + o_r1);
    float*  attnum = (float*) (w + o_r1);
    float*  pre1   = (float*) (w + o_pre1);
    __bf16* qbf    = (__bf16*)(w + o_pre1);
    __bf16* aggbf  = (__bf16*)(w + o_pre1 + (size_t)NN * 256 * 2);
    __bf16* obbf   = (__bf16*)(w + o_pre1 + (size_t)NN * 256 * 3);
    float*  qkv    = (float*) (w + o_qkv);
    float*  agg    = (float*) (w + o_qkv);
    float*  h2pre  = (float*) (w + o_qkv + (size_t)NN * HH * 4);
    float*  ht     = (float*) (w + o_ht);
    __bf16* htbf   = (__bf16*)(w + o_htbf);
    float*  x1     = (float*) (w + o_x1);
    float*  hg     = (float*) (w + o_hg);
    float*  deg    = (float*) (w + o_deg);
    __bf16* wcat   = (__bf16*)(w + o_wcat);
    __bf16* wqkv   = (__bf16*)(w + o_wqkv);
    __bf16* wconv  = (__bf16*)(w + o_wcv);
    __bf16* wfc    = (__bf16*)(w + o_wfc);
    float*  bcat   = (float*) (w + o_bcat);
    float*  bqkv   = (float*) (w + o_bqkv);
    float*  kvs    = (float*) (w + o_kvs);
    __bf16* kvsbf  = (__bf16*)(w + o_kvsb);
    float*  ksum   = (float*) (w + o_ksum);
    float*  sums   = (float*) (w + o_sums);

    const int STRIPS = NN / 16;                      // 6250
    const dim3 GB(256);
    const int GS = 4096;                             // grid-stride blocks for elementwise

    // 1) weight prep (bf16, transposed) + concat biases
    prep_weights_kernel<<<996, 256, 0, stream>>>(t_fc_w, g_fc_w, wq_w, wk_w, wv_w, conv_w, fc_w,
                                                 t_fc_b, g_fc_b, wq_b, wk_b, wv_b,
                                                 wcat, wqkv, wconv, wfc, bcat, bqkv);
    // 2) x -> bf16
    convert_bf16_kernel<<<GS, GB, 0, stream>>>(x, xbf, (long)NN * FF);
    // 3) fused input GEMM: [N,512] x [512,256] -> pre1 (t | g halves)
    gemm_bf16<<<dim3((STRIPS + 7) / 8, 4), 256, 0, stream>>>(xbf, FF, wcat, FF, bcat, pre1, 256, STRIPS, FF);
    // 4) LN0+ReLU -> ht (+bf16 copy), BN0+ReLU -> hg
    ln0_relu_kernel<<<NN, 128, 0, stream>>>(pre1, ln0_g, ln0_b, ht, htbf);
    bn0_relu_kernel<<<GS, GB, 0, stream>>>(pre1, bn0_g, bn0_b, hg, (long)NN * HH);
    // 5) QKV GEMM: [N,128] x [128,768]
    gemm_bf16<<<dim3((STRIPS + 7) / 8, 12), 256, 0, stream>>>(htbf, HH, wqkv, HH, bqkv, qkv, 768, STRIPS, HH);
    // 6) reductions: |q|^2,|k|^2 ; column sums of k ; kvs = k^T v (split-L + atomics)
    zero_f32_kernel<<<1, 64, 0, stream>>>(sums, 2);
    zero_f32_kernel<<<1, 256, 0, stream>>>(ksum, 256);
    zero_f32_kernel<<<128, 256, 0, stream>>>(kvs, 2 * 128 * 128);
    sumsq_kernel<<<GS, GB, 0, stream>>>(qkv, sums, (long)NN * 256);
    ksum_kernel<<<256, 256, 0, stream>>>(qkv, ksum);
    kvs_partial_kernel<<<dim3(64, 2, 40), 256, 0, stream>>>(qkv, kvs, 2500);
    kvs_t_bf_kernel<<<128, 256, 0, stream>>>(kvs, kvsbf);
    // 7) attention numerators: per-head [N,128] x [128,128] WMMA GEMMs
    qbf_kernel<<<GS, GB, 0, stream>>>(qkv, qbf, (long)NN * 256);
    gemm_bf16<<<dim3((STRIPS + 7) / 8, 2), 256, 0, stream>>>(qbf, 256, kvsbf, 128, nullptr, attnum, 256, STRIPS, HH);
    gemm_bf16<<<dim3((STRIPS + 7) / 8, 2), 256, 0, stream>>>(qbf + 128, 256, kvsbf + 16384, 128, nullptr, attnum + 128, 256, STRIPS, HH);
    // 8) attention epilogue + residual + LN1 + ReLU -> x1
    attn_post_kernel<<<NN, 128, 0, stream>>>(qkv, attnum, ht, ksum, sums, ln1_g, ln1_b, x1);
    // 9) graph conv: degree, normalized scatter-add, conv GEMM, BN1+ReLU+residual
    zero_f32_kernel<<<512, 256, 0, stream>>>(deg, NN);
    deg_kernel<<<(EE + 255) / 256, 256, 0, stream>>>(ei, deg);
    zero_f32_kernel<<<GS, GB, 0, stream>>>(agg, (long)NN * HH);
    scatter_kernel<<<(int)(((long)EE * 32 + 255) / 256), 256, 0, stream>>>(ei, hg, deg, agg);
    convert_bf16_kernel<<<GS, GB, 0, stream>>>(agg, aggbf, (long)NN * HH);
    gemm_bf16<<<dim3((STRIPS + 7) / 8, 2), 256, 0, stream>>>(aggbf, HH, wconv, HH, conv_b, h2pre, HH, STRIPS, HH);
    gc_combine_kernel<<<GS, GB, 0, stream>>>(h2pre, hg, bn1_g, bn1_b, x1, obbf, (long)NN * HH);
    // 10) output GEMM: [N,128] x [128,64] -> d_out
    gemm_bf16<<<dim3((STRIPS + 7) / 8, 1), 256, 0, stream>>>(obbf, HH, wfc, HH, fc_b, out, OC, STRIPS, HH);
}
